// _MultimodalCrossLayer_23313082483322
// MI455X (gfx1250) — matrix operand order repaired and checked by reading, not compiled, not run on hardware
//
#include <hip/hip_runtime.h>
#include <hip/hip_bf16.h>
#include <math.h>

// ---------------------------------------------------------------------------
// Types
// ---------------------------------------------------------------------------
typedef _Float16 h16;
typedef __attribute__((ext_vector_type(16))) _Float16 v16h;
typedef __attribute__((ext_vector_type(8)))  _Float16 v8h;
typedef __attribute__((ext_vector_type(8)))  float    v8f;
typedef __attribute__((ext_vector_type(4)))  int      v4i;

#define DEVI __device__ __forceinline__

static constexpr int BATCH = 8;
static constexpr int SLEN  = 512;
static constexpr int DIM   = 1024;
static constexpr int NHEAD = 16;
static constexpr int DHEAD = 64;
static constexpr int MLPD  = 4096;

// ---------------------------------------------------------------------------
// CDNA5 async global->LDS staging (GLOBAL_LOAD_ASYNC_TO_LDS_B128, ASYNCcnt),
// with a synchronous fallback if the builtin is not declared by this clang.
// Builtin prototype (probe-confirmed): (v4i AS1*, v4i AS3*, imm, imm).
// ---------------------------------------------------------------------------
#if __has_builtin(__builtin_amdgcn_global_load_async_to_lds_b128)
#define HAS_ASYNC_LDS 1
#else
#define HAS_ASYNC_LDS 0
#endif

DEVI void stage16(const h16* g, h16* l) {
#if HAS_ASYNC_LDS
  typedef __attribute__((address_space(1))) v4i GAS;
  typedef __attribute__((address_space(3))) v4i LAS;
  __builtin_amdgcn_global_load_async_to_lds_b128((GAS*)g, (LAS*)l, 0, 0);
#else
  *(v8h*)l = *(const v8h*)g;
#endif
}
DEVI void stage_wait() {
#if HAS_ASYNC_LDS
  asm volatile("s_wait_asynccnt 0x0" ::: "memory");
#endif
}

// ---------------------------------------------------------------------------
// Butterfly reductions over 16-lane groups via DS_SWIZZLE (group-of-32 xor
// mode: new_lane = ((lane & 0x1F) | 0) ^ mask) — no bounds-check VALU chain.
// ---------------------------------------------------------------------------
#if __has_builtin(__builtin_amdgcn_ds_swizzle)
#define HAS_SWZ 1
#else
#define HAS_SWZ 0
#endif

template <int M>
DEVI float swz_xor(float v) {
#if HAS_SWZ
  return __int_as_float(__builtin_amdgcn_ds_swizzle(__float_as_int(v), (M << 10) | 0x1F));
#else
  return __shfl_xor(v, M, 32);
#endif
}
DEVI float wave_max16(float v) {
  v = fmaxf(v, swz_xor<1>(v));
  v = fmaxf(v, swz_xor<2>(v));
  v = fmaxf(v, swz_xor<4>(v));
  v = fmaxf(v, swz_xor<8>(v));
  return v;
}
DEVI float wave_sum16(float v) {
  v += swz_xor<1>(v);
  v += swz_xor<2>(v);
  v += swz_xor<4>(v);
  v += swz_xor<8>(v);
  return v;
}

// ---------------------------------------------------------------------------
// Fragment helpers (CDNA5 WMMA 16x16x32 f16, wave32 layouts per ISA 7.12.2)
// A operand: lane holds row M = lane%16; element i <-> K = (i&7) + (i>>3)*16 + (lane>=16)*8
// B operand: lane holds col N = lane%16; element i <-> K = (i&7) + (i>>3)*16 + (lane>=16)*8
// C/D      : lanes 0-15: N=lane, elem r -> M=r ; lanes 16-31: N=lane-16, M=r+8
// ---------------------------------------------------------------------------
DEVI v16h make_frag(v8h lo, v8h hi) {
  v16h f;
#pragma unroll
  for (int e = 0; e < 8; ++e) { f[e] = lo[e]; f[e + 8] = hi[e]; }
  return f;
}
DEVI v16h load_afrag(const h16* rowptr, int hb8) {
  v8h lo = *(const v8h*)(rowptr + hb8);
  v8h hi = *(const v8h*)(rowptr + 16 + hb8);
  return make_frag(lo, hi);
}
DEVI v16h load_bfrag(const h16* rowptr, int hb16) {
  v8h lo = *(const v8h*)(rowptr + hb16 / 2);
  v8h hi = *(const v8h*)(rowptr + 16 + hb16 / 2);
  return make_frag(lo, hi);
}

// ---------------------------------------------------------------------------
// Elementwise kernels
// ---------------------------------------------------------------------------
__global__ void k_f32_to_h16(const float* __restrict__ in, h16* __restrict__ out, int n) {
  int i = blockIdx.x * blockDim.x + threadIdx.x;
  if (i < n) out[i] = (h16)in[i];
}

// in: f32 [K,N] row-major -> out: f16 [N,K] row-major (weights pre-transposed)
__global__ __launch_bounds__(256) void k_transpose_h16(const float* __restrict__ in,
                                                       h16* __restrict__ out, int K, int N) {
  __shared__ float tile[32][33];
  int k0 = blockIdx.y * 32, n0 = blockIdx.x * 32;
  int tx = threadIdx.x & 31, ty = threadIdx.x >> 5; // 8 rows of 32
  for (int r = ty; r < 32; r += 8) tile[r][tx] = in[(size_t)(k0 + r) * N + (n0 + tx)];
  __syncthreads();
  for (int r = ty; r < 32; r += 8) out[(size_t)(n0 + r) * K + (k0 + tx)] = (h16)tile[tx][r];
}

// copy f16 [rows,cols] into a wider matrix (row stride dstride) at column 0
__global__ void k_copy_stride(const h16* __restrict__ src, h16* __restrict__ dst,
                              int rows, int cols, int dstride) {
  int i = blockIdx.x * blockDim.x + threadIdx.x;
  if (i >= rows * cols) return;
  int r = i / cols, c = i - r * cols;
  dst[(size_t)r * dstride + c] = src[(size_t)i];
}

// h = x + gate * attn ; scatter into fused layout [b, part*512 + s, :]
__global__ void k_gate_combine(const float* __restrict__ x, const float* __restrict__ gate,
                               const h16* __restrict__ cat /* [Ns,2048], attn at col 1024 */,
                               float* __restrict__ fusedf, h16* __restrict__ fused16, int part) {
  int i = blockIdx.x * blockDim.x + threadIdx.x;
  if (i >= 4096 * 1024) return;
  int r = i >> 10, c = i & 1023;
  int bb = r >> 9, s = r & 511;
  size_t fr = (((size_t)bb * 1024) + part * 512 + s) * 1024 + c;
  float a = (float)cat[(size_t)r * 2048 + 1024 + c];
  float v = x[i] + gate[i] * a;
  fusedf[fr] = v;
  fused16[fr] = (h16)v;
}

// LayerNorm over D=1024 of (a+b); optional f32/f16 outputs, optional final split write
__global__ __launch_bounds__(256) void k_ln(const float* __restrict__ a, const float* __restrict__ b,
                                            const float* __restrict__ g, const float* __restrict__ be,
                                            float* __restrict__ outf, h16* __restrict__ out16,
                                            float* __restrict__ split0, float* __restrict__ split1) {
  __shared__ float rs[256], rss[256];
  const int tid = threadIdx.x;
  const size_t row = blockIdx.x;
  const float* ar = a + row * DIM;
  const float* br = b + row * DIM;
  float x[4], s = 0.f, ss = 0.f;
#pragma unroll
  for (int j = 0; j < 4; ++j) {
    int c = tid + j * 256;
    float v = ar[c] + br[c];
    x[j] = v; s += v; ss += v * v;
  }
  rs[tid] = s; rss[tid] = ss; __syncthreads();
  for (int off = 128; off > 0; off >>= 1) {
    if (tid < off) { rs[tid] += rs[tid + off]; rss[tid] += rss[tid + off]; }
    __syncthreads();
  }
  float mean = rs[0] * (1.0f / DIM);
  float var  = rss[0] * (1.0f / DIM) - mean * mean;
  float inv  = rsqrtf(var + 1e-12f);
#pragma unroll
  for (int j = 0; j < 4; ++j) {
    int c = tid + j * 256;
    float y = (x[j] - mean) * inv * g[c] + be[c];
    if (outf)  outf[row * DIM + c] = y;
    if (out16) out16[row * DIM + c] = (h16)y;
    if (split0) {
      int bb = (int)(row >> 10), sq = (int)(row & 1023);
      if (sq < 512) split0[(((size_t)bb * 512) + sq) * DIM + c] = y;
      else          split1[(((size_t)bb * 512) + sq - 512) * DIM + c] = y;
    }
  }
}

// ---------------------------------------------------------------------------
// WMMA GEMM:  C[M,N] = A[M,K] * B[K,N] + bias ; Bt is B transposed ([N,K] f16).
// 256 threads = 8 waves; block tile 128x128, BK=64 (two 32-K WMMA steps per
// staging round -> 16 WMMAs per wave between barrier pairs).
// Waves in a 4(row)x2(col) grid; each wave owns a 32x64 sub-tile.
// ---------------------------------------------------------------------------
enum { EPI_NONE = 0, EPI_SIGMOID = 1, EPI_GELU = 2 };

template <int EPI>
__global__ __launch_bounds__(256)
void k_gemm(const h16* __restrict__ A, const h16* __restrict__ Bt,
            const float* __restrict__ bias,
            float* __restrict__ Cf, h16* __restrict__ Cb,
            int M, int N, int K) {
  __shared__ alignas(16) h16 As[128][72]; // row stride 144B (16B multiple)
  __shared__ alignas(16) h16 Bs[128][72]; // Bs[n][k]
  const int tid = threadIdx.x;
  const int wave = tid >> 5, lane = tid & 31;
  const int lm = lane & 15, hb = lane >> 4;
  const int wr = wave & 3, wc = wave >> 2; // 4 row-groups x 2 col-groups
  const int row0 = blockIdx.y * 128, col0 = blockIdx.x * 128;

  v8f acc[2][4];
#pragma unroll
  for (int i = 0; i < 2; ++i)
#pragma unroll
    for (int j = 0; j < 4; ++j)
#pragma unroll
      for (int e = 0; e < 8; ++e) acc[i][j][e] = 0.f;

  for (int kk = 0; kk < K; kk += 64) {
    __syncthreads();
    // stage A[128x64] and Bt[128x64]: 1024 16B chunks each way, 4 per thread
#pragma unroll
    for (int c = tid; c < 1024; c += 256) {
      int r = c >> 3, cc = (c & 7) << 3;
      stage16(&A [(size_t)(row0 + r) * K + kk + cc], &As[r][cc]);
      stage16(&Bt[(size_t)(col0 + r) * K + kk + cc], &Bs[r][cc]);
    }
    stage_wait();
    __syncthreads();

#pragma unroll
    for (int ks = 0; ks < 64; ks += 32) {
      v16h af[2], bf[4];
#pragma unroll
      for (int i = 0; i < 2; ++i) af[i] = load_afrag(&As[wr * 32 + i * 16 + lm][ks], hb * 8);
#pragma unroll
      for (int j = 0; j < 4; ++j) bf[j] = load_bfrag(&Bs[wc * 64 + j * 16 + lm][ks], hb * 16);
#pragma unroll
      for (int i = 0; i < 2; ++i)
#pragma unroll
        for (int j = 0; j < 4; ++j)
          acc[i][j] = __builtin_amdgcn_wmma_f32_16x16x32_f16(false, af[i], false, bf[j],
                                                             (short)0, acc[i][j], false, false);
    }
  }

#pragma unroll
  for (int j = 0; j < 4; ++j) {
    int col = col0 + wc * 64 + j * 16 + lm;
    float bv = bias ? bias[col] : 0.f;
#pragma unroll
    for (int i = 0; i < 2; ++i)
#pragma unroll
      for (int r = 0; r < 8; ++r) {
        int row = row0 + wr * 32 + i * 16 + r + hb * 8;
        float v = acc[i][j][r] + bv;
        if (EPI == EPI_SIGMOID) v = 1.f / (1.f + __expf(-v));
        if (EPI == EPI_GELU)    v = 0.5f * v * (1.f + erff(v * 0.70710678f));
        size_t idx = (size_t)row * N + col;
        if (Cf) Cf[idx] = v;
        if (Cb) Cb[idx] = (h16)v;
      }
  }
}

// ---------------------------------------------------------------------------
// Flash attention with optional split softmax (keys [0,seg) weight w0, [seg,klen) weight w1).
// Grid: (B*NHEAD, qlen/64); 128 threads = 4 waves; each wave owns a 16-query tile.
// ---------------------------------------------------------------------------
__global__ __launch_bounds__(128)
void k_attention(const h16* __restrict__ Q, const h16* __restrict__ K,
                 const h16* __restrict__ V, h16* __restrict__ O,
                 int ostride, int ocol, int qlen, int klen, int seg,
                 const float* __restrict__ w0p, const float* __restrict__ w1p) {
  __shared__ alignas(16) h16 Ks[32][72];     // [key][dh]   row 144B
  __shared__ alignas(16) h16 Vs[64][40];     // [dh][key]   row 80B
  __shared__ alignas(16) h16 Ps[4][16][40];  // per-wave P  row 80B

  const int tid = threadIdx.x;
  const int wave = tid >> 5, lane = tid & 31;
  const int lm = lane & 15, hb = lane >> 4;
  const int b = blockIdx.x / NHEAD, h = blockIdx.x % NHEAD;
  const int q0 = blockIdx.y * 64 + wave * 16;
  const float scale = 0.125f; // 1/sqrt(64)
  const float w0 = w0p ? *w0p : 1.0f;
  const float w1 = w1p ? *w1p : 0.0f;

  const h16* qrow = Q + (size_t)(b * qlen + q0 + lm) * DIM + h * DHEAD;
  v16h aq0 = load_afrag(qrow, hb * 8);
  v16h aq1 = load_afrag(qrow + 32, hb * 8);

  v8f res[4];
#pragma unroll
  for (int j = 0; j < 4; ++j)
#pragma unroll
    for (int e = 0; e < 8; ++e) res[j][e] = 0.f;

  for (int si = 0; si < 2; ++si) {
    const int kbeg = si ? seg : 0;
    const int kend = si ? klen : seg;
    if (kbeg >= kend) continue; // uniform per block
    const float wseg = si ? w1 : w0;

    v8f o[4];
#pragma unroll
    for (int j = 0; j < 4; ++j)
#pragma unroll
      for (int e = 0; e < 8; ++e) o[j][e] = 0.f;
    float mx[8], ls[8];
#pragma unroll
    for (int r = 0; r < 8; ++r) { mx[r] = -INFINITY; ls[r] = 0.f; }

    for (int kt = kbeg; kt < kend; kt += 32) {
      __syncthreads();
      // stage K tile [32 keys][64 dh] (async path) + V tile transposed [64 dh][32 keys]
#pragma unroll
      for (int c = tid; c < 256; c += 128) {
        int r = c >> 3, cc = (c & 7) << 3;
        const size_t grow = (size_t)(b * klen + kt + r) * DIM + h * DHEAD + cc;
        stage16(&K[grow], &Ks[r][cc]);
        v8h vv = *(const v8h*)&V[grow];
#pragma unroll
        for (int e = 0; e < 8; ++e) Vs[cc + e][r] = vv[e];
      }
      stage_wait();
      __syncthreads();

      // scores S = Q * K^T  (16 queries x 32 keys), two 16-col WMMA tiles
      v8f s[2];
#pragma unroll
      for (int jn = 0; jn < 2; ++jn) {
        v8f z;
#pragma unroll
        for (int e = 0; e < 8; ++e) z[e] = 0.f;
        const h16* krow = &Ks[jn * 16 + lm][0];
        v16h bk0 = load_bfrag(krow, hb * 16);
        v16h bk1 = load_bfrag(krow + 32, hb * 16);
        z = __builtin_amdgcn_wmma_f32_16x16x32_f16(false, aq0, false, bk0, (short)0, z, false, false);
        z = __builtin_amdgcn_wmma_f32_16x16x32_f16(false, aq1, false, bk1, (short)0, z, false, false);
        s[jn] = z;
      }

      // online softmax update (row = r + hb*8, cols spread over 16 lanes)
      float corr[8];
#pragma unroll
      for (int r = 0; r < 8; ++r) {
        float v0 = s[0][r] * scale, v1 = s[1][r] * scale;
        float rmax = wave_max16(fmaxf(v0, v1));
        float mnew = fmaxf(mx[r], rmax);
        float p0 = __expf(v0 - mnew), p1 = __expf(v1 - mnew);
        corr[r] = __expf(mx[r] - mnew);
        float rsum = wave_sum16(p0 + p1);
        ls[r] = ls[r] * corr[r] + rsum;
        mx[r] = mnew;
        Ps[wave][r + hb * 8][lm]      = (h16)p0;
        Ps[wave][r + hb * 8][16 + lm] = (h16)p1;
      }
#pragma unroll
      for (int j = 0; j < 4; ++j)
#pragma unroll
        for (int r = 0; r < 8; ++r) o[j][r] *= corr[r];

      __syncthreads();

      // ctx += P (16x32, A operand) * V (32x16 per dh tile, B operand)
      v16h pf = load_afrag(&Ps[wave][lm][0], hb * 8);
#pragma unroll
      for (int jd = 0; jd < 4; ++jd) {
        v16h bv = load_bfrag(&Vs[jd * 16 + lm][0], hb * 16);
        o[jd] = __builtin_amdgcn_wmma_f32_16x16x32_f16(false, pf, false, bv,
                                                       (short)0, o[jd], false, false);
      }
    }

    // fold normalized segment into result with its weight
#pragma unroll
    for (int r = 0; r < 8; ++r) {
      float f = wseg / ls[r];
#pragma unroll
      for (int j = 0; j < 4; ++j) res[j][r] += o[j][r] * f;
    }
  }

#pragma unroll
  for (int jd = 0; jd < 4; ++jd)
#pragma unroll
    for (int r = 0; r < 8; ++r) {
      size_t row = (size_t)b * qlen + q0 + r + hb * 8;
      O[row * ostride + ocol + h * DHEAD + jd * 16 + lm] = (h16)res[jd][r];
    }
}

// ---------------------------------------------------------------------------
// Host orchestration
// ---------------------------------------------------------------------------
extern "C" void kernel_launch(void* const* d_in, const int* in_sizes, int n_in,
                              void* d_out, int out_size, void* d_ws, size_t ws_size,
                              hipStream_t stream) {
  const float* source_x = (const float*)d_in[0];
  const float* target_x = (const float*)d_in[1];
  const float* Wq = (const float*)d_in[2];  const float* bq = (const float*)d_in[3];
  const float* Wk = (const float*)d_in[4];  const float* bk = (const float*)d_in[5];
  const float* Wv = (const float*)d_in[6];  const float* bv = (const float*)d_in[7];
  const float* Wd = (const float*)d_in[8];  const float* bd = (const float*)d_in[9];
  const float* ln1_g = (const float*)d_in[10]; const float* ln1_b = (const float*)d_in[11];
  const float* w_s = (const float*)d_in[12];   const float* w_t = (const float*)d_in[13];
  const float* Wgs = (const float*)d_in[14];   const float* bgs = (const float*)d_in[15];
  const float* Wgt = (const float*)d_in[16];   const float* bgt = (const float*)d_in[17];
  const float* W1 = (const float*)d_in[18];    const float* b1 = (const float*)d_in[19];
  const float* W2 = (const float*)d_in[20];    const float* b2 = (const float*)d_in[21];
  const float* ln2_g = (const float*)d_in[22]; const float* ln2_b = (const float*)d_in[23];

  constexpr int Ns = BATCH * SLEN;     // 4096 source (= target) tokens
  constexpr int Nf = BATCH * 2 * SLEN; // 8192 fused tokens
  constexpr size_t MB = 1024 * 1024;

  char* wp = (char*)d_ws;
  auto alloc = [&](size_t bytes) -> char* {
    char* p = wp; wp += (bytes + 255) & ~(size_t)255; return p;
  };

  h16* xs16   = (h16*)alloc((size_t)Ns * DIM * 2);          // 8 MB
  h16* xt16   = (h16*)alloc((size_t)Ns * DIM * 2);          // 8 MB
  h16* Wq_t   = (h16*)alloc((size_t)DIM * DIM * 2);
  h16* Wk_t   = (h16*)alloc((size_t)DIM * DIM * 2);
  h16* Wv_t   = (h16*)alloc((size_t)DIM * DIM * 2);
  h16* Wd_t   = (h16*)alloc((size_t)DIM * DIM * 2);
  h16* Wgs_t  = (h16*)alloc((size_t)DIM * 2 * DIM * 2);
  h16* Wgt_t  = (h16*)alloc((size_t)DIM * 2 * DIM * 2);
  h16* W1_t   = (h16*)alloc((size_t)MLPD * DIM * 2);
  h16* W2_t   = (h16*)alloc((size_t)DIM * MLPD * 2);
  h16* Qs     = (h16*)alloc((size_t)Ns * DIM * 2);          // 6 x 8 MB contiguous
  h16* Ksrc   = (h16*)alloc((size_t)Ns * DIM * 2);
  h16* Vsrc   = (h16*)alloc((size_t)Ns * DIM * 2);
  h16* Qt     = (h16*)alloc((size_t)Ns * DIM * 2);
  h16* Kt     = (h16*)alloc((size_t)Ns * DIM * 2);
  h16* Vt     = (h16*)alloc((size_t)Ns * DIM * 2);
  h16* cat_s  = (h16*)alloc((size_t)Ns * 2 * DIM * 2);      // 16 MB
  h16* cat_t  = (h16*)alloc((size_t)Ns * 2 * DIM * 2);      // 16 MB
  float* gate_s = (float*)alloc((size_t)Ns * DIM * 4);      // 16 MB
  float* gate_t = (float*)alloc((size_t)Ns * DIM * 4);      // 16 MB
  float* fusedf = (float*)alloc((size_t)Nf * DIM * 4);      // 32 MB
  h16* fused16  = (h16*)alloc((size_t)Nf * DIM * 2);        // 16 MB
  float* hidf   = (float*)alloc((size_t)Nf * DIM * 4);      // 32 MB
  h16* ff1_16   = (h16*)alloc((size_t)Nf * MLPD * 2);       // 64 MB
  (void)in_sizes; (void)n_in; (void)out_size; (void)ws_size;

  // Aliased reuse of dead buffers:
  h16*   Qf    = Qs;                                  // QKV-cross dead after cross attn
  h16*   Kf    = (h16*)((char*)Qs + 16 * MB);
  h16*   Vf    = (h16*)((char*)Qs + 32 * MB);
  h16*   ctx16 = cat_s;                               // cat_s dead after gates/combine
  float* attnf = gate_s;                              // gate_s+gate_t (32 MB) dead after combine
  h16*   hid16 = cat_t;                               // cat_t dead after combine
  float* ff2f  = fusedf;                              // fusedf dead after LN1

  auto gemm = [&](const h16* A, const h16* Bt, const float* bias, float* Cf, h16* Cb,
                  int M, int N, int K, int epi) {
    dim3 g(N / 128, M / 128);
    if (epi == EPI_SIGMOID)
      k_gemm<EPI_SIGMOID><<<g, 256, 0, stream>>>(A, Bt, bias, Cf, Cb, M, N, K);
    else if (epi == EPI_GELU)
      k_gemm<EPI_GELU><<<g, 256, 0, stream>>>(A, Bt, bias, Cf, Cb, M, N, K);
    else
      k_gemm<EPI_NONE><<<g, 256, 0, stream>>>(A, Bt, bias, Cf, Cb, M, N, K);
  };

  const int nAct = Ns * DIM; // 4 Mi elements

  // 1) precision conversion + weight transposition
  k_f32_to_h16<<<nAct / 256, 256, 0, stream>>>(source_x, xs16, nAct);
  k_f32_to_h16<<<nAct / 256, 256, 0, stream>>>(target_x, xt16, nAct);
  k_transpose_h16<<<dim3(DIM / 32, DIM / 32), 256, 0, stream>>>(Wq, Wq_t, DIM, DIM);
  k_transpose_h16<<<dim3(DIM / 32, DIM / 32), 256, 0, stream>>>(Wk, Wk_t, DIM, DIM);
  k_transpose_h16<<<dim3(DIM / 32, DIM / 32), 256, 0, stream>>>(Wv, Wv_t, DIM, DIM);
  k_transpose_h16<<<dim3(DIM / 32, DIM / 32), 256, 0, stream>>>(Wd, Wd_t, DIM, DIM);
  k_transpose_h16<<<dim3(DIM / 32, 2 * DIM / 32), 256, 0, stream>>>(Wgs, Wgs_t, 2 * DIM, DIM);
  k_transpose_h16<<<dim3(DIM / 32, 2 * DIM / 32), 256, 0, stream>>>(Wgt, Wgt_t, 2 * DIM, DIM);
  k_transpose_h16<<<dim3(MLPD / 32, DIM / 32), 256, 0, stream>>>(W1, W1_t, DIM, MLPD);
  k_transpose_h16<<<dim3(DIM / 32, MLPD / 32), 256, 0, stream>>>(W2, W2_t, MLPD, DIM);

  // 2) cross-attention projections
  gemm(xs16, Wq_t, bq, nullptr, Qs,   Ns, DIM, DIM, EPI_NONE);
  gemm(xs16, Wk_t, bk, nullptr, Ksrc, Ns, DIM, DIM, EPI_NONE);
  gemm(xs16, Wv_t, bv, nullptr, Vsrc, Ns, DIM, DIM, EPI_NONE);
  gemm(xt16, Wq_t, bq, nullptr, Qt,   Ns, DIM, DIM, EPI_NONE);
  gemm(xt16, Wk_t, bk, nullptr, Kt,   Ns, DIM, DIM, EPI_NONE);
  gemm(xt16, Wv_t, bv, nullptr, Vt,   Ns, DIM, DIM, EPI_NONE);

  // 3) cross attentions, outputs written into the concat buffers' second halves
  k_copy_stride<<<nAct / 256, 256, 0, stream>>>(xs16, cat_s, Ns, DIM, 2 * DIM);
  k_copy_stride<<<nAct / 256, 256, 0, stream>>>(xt16, cat_t, Ns, DIM, 2 * DIM);
  k_attention<<<dim3(BATCH * NHEAD, SLEN / 64), 128, 0, stream>>>(
      Qs, Kt, Vt, cat_s, 2 * DIM, DIM, SLEN, SLEN, SLEN, nullptr, nullptr);
  k_attention<<<dim3(BATCH * NHEAD, SLEN / 64), 128, 0, stream>>>(
      Qt, Ksrc, Vsrc, cat_t, 2 * DIM, DIM, SLEN, SLEN, SLEN, nullptr, nullptr);

  // 4) gates + combine into fused representation
  gemm(cat_s, Wgs_t, bgs, gate_s, nullptr, Ns, DIM, 2 * DIM, EPI_SIGMOID);
  gemm(cat_t, Wgt_t, bgt, gate_t, nullptr, Ns, DIM, 2 * DIM, EPI_SIGMOID);
  k_gate_combine<<<nAct / 256, 256, 0, stream>>>(source_x, gate_s, cat_s, fusedf, fused16, 0);
  k_gate_combine<<<nAct / 256, 256, 0, stream>>>(target_x, gate_t, cat_t, fusedf, fused16, 1);

  // 5) fused self-attention with split softmax (w_s / w_t)
  gemm(fused16, Wq_t, bq, nullptr, Qf, Nf, DIM, DIM, EPI_NONE);
  gemm(fused16, Wk_t, bk, nullptr, Kf, Nf, DIM, DIM, EPI_NONE);
  gemm(fused16, Wv_t, bv, nullptr, Vf, Nf, DIM, DIM, EPI_NONE);
  k_attention<<<dim3(BATCH * NHEAD, (2 * SLEN) / 64), 128, 0, stream>>>(
      Qf, Kf, Vf, ctx16, DIM, 0, 2 * SLEN, 2 * SLEN, SLEN, w_s, w_t);
  gemm(ctx16, Wd_t, bd, attnf, nullptr, Nf, DIM, DIM, EPI_NONE);

  // 6) LN1 -> MLP -> LN2 (final writes directly into split outputs)
  k_ln<<<Nf, 256, 0, stream>>>(attnf, fusedf, ln1_g, ln1_b, hidf, hid16, nullptr, nullptr);
  gemm(hid16, W1_t, b1, nullptr, ff1_16, Nf, MLPD, DIM, EPI_GELU);
  gemm(ff1_16, W2_t, b2, ff2f, nullptr, Nf, DIM, MLPD, EPI_NONE);
  float* out0 = (float*)d_out;
  float* out1 = out0 + (size_t)BATCH * SLEN * DIM;
  k_ln<<<Nf, 256, 0, stream>>>(ff2f, hidf, ln2_g, ln2_b, nullptr, nullptr, out0, out1);
}
